// GraphUnet_9139690406274
// MI455X (gfx1250) — 8-bit matrix operand order repaired and checked by reading, compiled, not run on hardware
//
#include <hip/hip_runtime.h>
#include <hip/hip_bf16.h>
#include <cstddef>

#define GN   4096
#define GD   512
#define GN1  3276
#define GN2  1965

typedef __attribute__((ext_vector_type(2))) float v2f;
typedef __attribute__((ext_vector_type(8))) float v8f;
typedef __attribute__((ext_vector_type(8))) int   v8i;

__device__ __forceinline__ void gemm_epilogue(
    const v8f& acc, int m_base, int n_out, int half,
    const float* __restrict__ preAdd, const float* __restrict__ bias,
    const float* __restrict__ postAdd, float* __restrict__ Out,
    int M, int N, int relu) {
#pragma unroll
  for (int r = 0; r < 8; ++r) {
    const int m_out = m_base + half * 8 + r;
    if (m_out < M && n_out < N) {
      float v = acc[r];
      const size_t o = (size_t)m_out * N + n_out;
      if (preAdd)  v += preAdd[o];
      if (bias)    v += bias[n_out];
      if (relu)    v  = fmaxf(v, 0.0f);
      if (postAdd) v += postAdd[o];
      Out[o] = v;
    }
  }
}

__global__ __launch_bounds__(256)
void gemm_f32_wmma(const float* __restrict__ A, const float* __restrict__ B,
                   const float* __restrict__ preAdd, const float* __restrict__ bias,
                   const float* __restrict__ postAdd, float* __restrict__ Out,
                   int M, int N, int K, int transA, int relu) {
  const int tilesX = (N + 63) >> 6;
  const int tilesY = (M + 15) >> 4;
  const long tiles = (long)tilesX * tilesY;
  const long wave  = (long)blockIdx.x * 8 + threadIdx.y;
  if (wave >= tiles) return;
  const int tm    = (int)(wave / tilesX);
  const int tn    = (int)(wave % tilesX);
  const int lane  = threadIdx.x;
  const int l15   = lane & 15;
  const int half  = lane >> 4;
  const int m     = tm * 16 + l15;
  const int nbase = tn * 64;
  const int Kmain = K & ~3;
  const bool fast = (tm * 16 + 16 <= M) && (nbase + 64 <= N);
  const size_t aStep = transA ? (size_t)M : 1;

  v8f acc0 = {}, acc1 = {}, acc2 = {}, acc3 = {};

  if (fast) {
    const float* __restrict__ Arow = transA ? (A + m) : (A + (size_t)m * K);
#pragma unroll 2
    for (int k0 = 0; k0 < Kmain; k0 += 4) {
      const int ka = k0 + 2 * half;
      v2f a = { Arow[(size_t)ka * aStep], Arow[((size_t)ka + 1) * aStep] };
      const float* __restrict__ b0p = B + (size_t)ka * N + nbase + l15;
      const float* __restrict__ b1p = b0p + N;
      v2f b0 = { b0p[0],  b1p[0]  };
      v2f b1 = { b0p[16], b1p[16] };
      v2f b2 = { b0p[32], b1p[32] };
      v2f b3 = { b0p[48], b1p[48] };
      acc0 = __builtin_amdgcn_wmma_f32_16x16x4_f32(false, a, false, b0, (short)0, acc0, false, false);
      acc1 = __builtin_amdgcn_wmma_f32_16x16x4_f32(false, a, false, b1, (short)0, acc1, false, false);
      acc2 = __builtin_amdgcn_wmma_f32_16x16x4_f32(false, a, false, b2, (short)0, acc2, false, false);
      acc3 = __builtin_amdgcn_wmma_f32_16x16x4_f32(false, a, false, b3, (short)0, acc3, false, false);
    }
    if (Kmain < K) {
      const int ka  = Kmain + 2 * half;
      const int ka0 = min(ka, K - 1);
      const int ka1 = min(ka + 1, K - 1);
      const float km0 = (ka     < K) ? 1.0f : 0.0f;
      const float km1 = (ka + 1 < K) ? 1.0f : 0.0f;
      const float* __restrict__ Arow = transA ? (A + m) : (A + (size_t)m * K);
      v2f a = { Arow[(size_t)ka0 * aStep] * km0, Arow[(size_t)ka1 * aStep] * km1 };
      const float* __restrict__ b0p = B + (size_t)ka0 * N + nbase + l15;
      const float* __restrict__ b1p = B + (size_t)ka1 * N + nbase + l15;
#pragma unroll
      for (int s = 0; s < 4; ++s) {
        v2f b = { b0p[s * 16], b1p[s * 16] };
        v8f& acc = (s == 0) ? acc0 : (s == 1) ? acc1 : (s == 2) ? acc2 : acc3;
        acc = __builtin_amdgcn_wmma_f32_16x16x4_f32(false, a, false, b, (short)0, acc, false, false);
      }
    }
  } else {
    const int   mc    = min(m, M - 1);
    const float mmask = (m < M) ? 1.0f : 0.0f;
    const float* __restrict__ Arow = transA ? (A + mc) : (A + (size_t)mc * K);
    for (int k0 = 0; k0 < K; k0 += 4) {
      const int ka  = k0 + 2 * half;
      const int ka0 = min(ka, K - 1);
      const int ka1 = min(ka + 1, K - 1);
      const float km0 = (ka     < K) ? mmask : 0.0f;
      const float km1 = (ka + 1 < K) ? mmask : 0.0f;
      v2f a = { Arow[(size_t)ka0 * aStep] * km0, Arow[(size_t)ka1 * aStep] * km1 };
#pragma unroll
      for (int s = 0; s < 4; ++s) {
        const int   n     = nbase + s * 16 + l15;
        const int   ncl   = min(n, N - 1);
        const float nmask = (n < N) ? 1.0f : 0.0f;
        v2f b;
        b.x = B[(size_t)ka0 * N + ncl] * nmask;
        b.y = B[(size_t)ka1 * N + ncl] * nmask;
        v8f& acc = (s == 0) ? acc0 : (s == 1) ? acc1 : (s == 2) ? acc2 : acc3;
        acc = __builtin_amdgcn_wmma_f32_16x16x4_f32(false, a, false, b, (short)0, acc, false, false);
      }
    }
  }

  gemm_epilogue(acc0, tm * 16, nbase      + l15, half, preAdd, bias, postAdd, Out, M, N, relu);
  gemm_epilogue(acc1, tm * 16, nbase + 16 + l15, half, preAdd, bias, postAdd, Out, M, N, relu);
  gemm_epilogue(acc2, tm * 16, nbase + 32 + l15, half, preAdd, bias, postAdd, Out, M, N, relu);
  gemm_epilogue(acc3, tm * 16, nbase + 48 + l15, half, preAdd, bias, postAdd, Out, M, N, relu);
}

__global__ __launch_bounds__(256)
void adj2_iu8_wmma(const unsigned char* __restrict__ A8, float* __restrict__ Out, int n) {
  const int tilesX = (n + 15) >> 4;
  const long tiles = (long)tilesX * tilesX;
  const long wave  = (long)blockIdx.x * 8 + threadIdx.y;
  if (wave >= tiles) return;
  const int tm   = (int)(wave / tilesX);
  const int tn   = (int)(wave % tilesX);
  const int lane = threadIdx.x;
  const int l15  = lane & 15;
  const int half = lane >> 4;
  const int m    = tm * 16 + l15;
  const int nc   = tn * 16 + l15;
  const int Kmain = n & ~63;
  const bool fast = (tm * 16 + 16 <= n) && (tn * 16 + 16 <= n);
  v8i acc = {};

  if (fast) {
    const unsigned char* __restrict__ Arow = A8 + (size_t)m  * n;
    const unsigned char* __restrict__ Brow = A8 + (size_t)nc * n;
    for (int k0 = 0; k0 < Kmain; k0 += 64) {
      v8i av, bv;
#pragma unroll
      for (int v = 0; v < 8; ++v) {
        const int kaBase = k0 + ((v & 3) << 2) + ((v >> 2) << 5) + (half << 4);
        av[v] = (int)*(const unsigned int*)(Arow + kaBase);
        const int kbBase = k0 + ((v & 3) << 2) + ((v >> 2) << 5) + (half << 4);
        bv[v] = (int)*(const unsigned int*)(Brow + kbBase);
      }
      acc = __builtin_amdgcn_wmma_i32_16x16x64_iu8(false, av, false, bv, acc, false, false);
    }
    if (Kmain < n) {
      v8i av, bv;
#pragma unroll
      for (int v = 0; v < 8; ++v) {
        const int kaBase = Kmain + ((v & 3) << 2) + ((v >> 2) << 5) + (half << 4);
        const int kbBase = Kmain + ((v & 3) << 2) + ((v >> 2) << 5) + (half << 4);
        unsigned int pa = 0, pb = 0;
#pragma unroll
        for (int j = 0; j < 4; ++j) {
          const int ka = kaBase + j, kb = kbBase + j;
          const unsigned int ba = (unsigned int)Arow[min(ka, n - 1)];
          const unsigned int bb = (unsigned int)Brow[min(kb, n - 1)];
          pa |= (ka < n ? ba : 0u) << (8 * j);
          pb |= (kb < n ? bb : 0u) << (8 * j);
        }
        av[v] = (int)pa; bv[v] = (int)pb;
      }
      acc = __builtin_amdgcn_wmma_i32_16x16x64_iu8(false, av, false, bv, acc, false, false);
    }
  } else {
    const unsigned char* __restrict__ Arow = A8 + (size_t)min(m,  n - 1) * n;
    const unsigned char* __restrict__ Brow = A8 + (size_t)min(nc, n - 1) * n;
    const bool mok = (m < n), nok = (nc < n);
    for (int k0 = 0; k0 < n; k0 += 64) {
      v8i av, bv;
#pragma unroll
      for (int v = 0; v < 8; ++v) {
        const int kaBase = k0 + ((v & 3) << 2) + ((v >> 2) << 5) + (half << 4);
        const int kbBase = k0 + ((v & 3) << 2) + ((v >> 2) << 5) + (half << 4);
        unsigned int pa = 0, pb = 0;
#pragma unroll
        for (int j = 0; j < 4; ++j) {
          const int ka = kaBase + j, kb = kbBase + j;
          const unsigned int ba = (unsigned int)Arow[min(ka, n - 1)];
          const unsigned int bb = (unsigned int)Brow[min(kb, n - 1)];
          pa |= ((mok && ka < n) ? ba : 0u) << (8 * j);
          pb |= ((nok && kb < n) ? bb : 0u) << (8 * j);
        }
        av[v] = (int)pa; bv[v] = (int)pb;
      }
      acc = __builtin_amdgcn_wmma_i32_16x16x64_iu8(false, av, false, bv, acc, false, false);
    }
  }

  for (int r = 0; r < 8; ++r) {
    const int m_out = tm * 16 + half * 8 + r;
    const int n_out = tn * 16 + l15;
    if (m_out < n && n_out < n)
      Out[(size_t)m_out * n + n_out] = (acc[r] > 0) ? 1.0f : 0.0f;
  }
}

__global__ void binarize_f32(const float* __restrict__ in, float* __restrict__ out, long tot) {
  const long i = (long)blockIdx.x * blockDim.x + threadIdx.x;
  if (i < tot) out[i] = (in[i] > 0.0f) ? 1.0f : 0.0f;
}
__global__ void binarize_u8(const float* __restrict__ in, unsigned char* __restrict__ out, long tot) {
  const long i = (long)blockIdx.x * blockDim.x + threadIdx.x;
  if (i < tot) out[i] = (in[i] > 0.0f) ? (unsigned char)1 : (unsigned char)0;
}
__global__ void zero_f32(float* __restrict__ out, long tot) {
  const long i = (long)blockIdx.x * blockDim.x + threadIdx.x;
  if (i < tot) out[i] = 0.0f;
}
__global__ void add_f32(const float* __restrict__ a, const float* __restrict__ b,
                        float* __restrict__ out, long tot) {
  const long i = (long)blockIdx.x * blockDim.x + threadIdx.x;
  if (i < tot) out[i] = a[i] + b[i];
}

__global__ void rowdot(const float* __restrict__ X, const float* __restrict__ w,
                       const float* __restrict__ b, float* __restrict__ out, int n, int d) {
  const int i = blockIdx.x * blockDim.x + threadIdx.x;
  if (i >= n) return;
  const float* row = X + (size_t)i * d;
  float acc = 0.0f;
  for (int j = 0; j < d; ++j) acc += row[j] * w[j];
  out[i] = acc + b[0];
}

__global__ __launch_bounds__(1024)
void dot_reduce(const float* __restrict__ a, const float* __restrict__ b,
                const float* __restrict__ ab, float* __restrict__ out, int n) {
  __shared__ float s[1024];
  float acc = 0.0f;
  for (int i = threadIdx.x; i < n; i += 1024) acc += a[i] * b[i];
  s[threadIdx.x] = acc;
  __syncthreads();
  for (int o = 512; o > 0; o >>= 1) {
    if (threadIdx.x < o) s[threadIdx.x] += s[threadIdx.x + o];
    __syncthreads();
  }
  if (threadIdx.x == 0) out[0] = s[0] + ab[0];
}

__global__ void sigmoid_combine(const float* __restrict__ fw, const float* __restrict__ c,
                                float* __restrict__ out, int n) {
  const int i = blockIdx.x * blockDim.x + threadIdx.x;
  if (i < n) out[i] = 1.0f / (1.0f + expf(-(fw[i] + c[0])));
}

__global__ __launch_bounds__(1024)
void topk_sort(const float* __restrict__ scores, int n, int k,
               float* __restrict__ vals, int* __restrict__ idxs) {
  __shared__ float sk[4096];
  __shared__ int   si[4096];
  const int tid = threadIdx.x;
  for (int i = tid; i < 4096; i += 1024) {
    sk[i] = (i < n) ? scores[i] : -__builtin_huge_valf();
    si[i] = i;
  }
  __syncthreads();
  for (int size = 2; size <= 4096; size <<= 1) {
    for (int stride = size >> 1; stride > 0; stride >>= 1) {
      for (int t = tid; t < 2048; t += 1024) {
        const int i = ((t / stride) * (stride << 1)) + (t % stride);
        const int j = i + stride;
        const float ki = sk[i], kj = sk[j];
        const int   ii = si[i], ij = si[j];
        const bool desc = ((i & size) == 0);
        const bool j_outranks_i = (kj > ki) || (kj == ki && ij < ii);
        const bool i_outranks_j = (ki > kj) || (ki == kj && ii < ij);
        const bool do_swap = desc ? j_outranks_i : i_outranks_j;
        if (do_swap) { sk[i] = kj; sk[j] = ki; si[i] = ij; si[j] = ii; }
      }
      __syncthreads();
    }
  }
  for (int i = tid; i < k; i += 1024) { vals[i] = sk[i]; idxs[i] = si[i]; }
}

__global__ void gather_scale(const float* __restrict__ h, const int* __restrict__ idx,
                             const float* __restrict__ vals, float* __restrict__ out, int d) {
  const int i = blockIdx.x;
  const int src = idx[i];
  const float s = vals[i];
  for (int j = threadIdx.x; j < d; j += blockDim.x)
    out[(size_t)i * d + j] = h[(size_t)src * d + j] * s;
}

__global__ void scatter_rows(const float* __restrict__ h, const int* __restrict__ idx,
                             float* __restrict__ out, int d) {
  const int i = blockIdx.x;
  const int dst = idx[i];
  for (int j = threadIdx.x; j < d; j += blockDim.x)
    out[(size_t)dst * d + j] = h[(size_t)i * d + j];
}

__global__ void gather_sub(const float* __restrict__ A2, const int* __restrict__ idx,
                           float* __restrict__ sub, int n, int k) {
  const long e = (long)blockIdx.x * blockDim.x + threadIdx.x;
  const long tot = (long)k * k;
  if (e >= tot) return;
  const int i = (int)(e / k), j = (int)(e % k);
  sub[e] = A2[(size_t)idx[i] * n + idx[j]];
}

__global__ void rowsum(const float* __restrict__ sub, float* __restrict__ deg, int k) {
  const int i = blockIdx.x * blockDim.x + threadIdx.x;
  if (i >= k) return;
  const float* row = sub + (size_t)i * k;
  float acc = 0.0f;
  for (int j = 0; j < k; ++j) acc += row[j];
  deg[i] = acc;
}

__global__ void colnorm(const float* __restrict__ sub, const float* __restrict__ deg,
                        float* __restrict__ g, int k) {
  const long e = (long)blockIdx.x * blockDim.x + threadIdx.x;
  const long tot = (long)k * k;
  if (e >= tot) return;
  g[e] = sub[e] / deg[e % k];
}

static inline int ew_blocks(long tot) { return (int)((tot + 255) / 256); }

static void launch_gemm(const float* A, int transA, const float* B,
                        const float* preAdd, const float* bias, int relu,
                        const float* postAdd, float* Out,
                        int M, int N, int K, hipStream_t s) {
  const long tiles = (long)((N + 63) >> 6) * ((M + 15) >> 4);
  const int blocks = (int)((tiles + 7) / 8);
  gemm_f32_wmma<<<blocks, dim3(32, 8), 0, s>>>(A, B, preAdd, bias, postAdd, Out,
                                               M, N, K, transA, relu);
}

static void launch_gin(const float* Abin, const float* h,
                       const float* w1, const float* b1,
                       const float* w2, const float* b2,
                       const float* skip, float* Z, float* T, float* out,
                       int n, hipStream_t s) {
  launch_gemm(Abin, 1, h,  h, nullptr, 0, nullptr, Z, n, GD, n, s);
  launch_gemm(Z, 0, w1, nullptr, b1, 1, nullptr, T, n, GD, GD, s);
  launch_gemm(T, 0, w2, nullptr, b2, 1, skip, out, n, GD, GD, s);
}

extern "C" void kernel_launch(void* const* d_in, const int* in_sizes, int n_in,
                              void* d_out, int out_size, void* d_ws, size_t ws_size,
                              hipStream_t stream) {
  const float* g   = (const float*)d_in[0];
  const float* h   = (const float*)d_in[1];
  const float* C0  = (const float*)d_in[2];
  const float* C1  = (const float*)d_in[3];
  const float* dw1 = (const float*)d_in[4];
  const float* db1 = (const float*)d_in[5];
  const float* dw2 = (const float*)d_in[6];
  const float* db2 = (const float*)d_in[7];
  const float* uw1 = (const float*)d_in[8];
  const float* ub1 = (const float*)d_in[9];
  const float* uw2 = (const float*)d_in[10];
  const float* ub2 = (const float*)d_in[11];
  const float* bw1 = (const float*)d_in[12];
  const float* bb1 = (const float*)d_in[13];
  const float* bw2 = (const float*)d_in[14];
  const float* bb2 = (const float*)d_in[15];
  const float* pwf = (const float*)d_in[16];
  const float* pbf = (const float*)d_in[17];
  const float* pws = (const float*)d_in[18];
  const float* pbs = (const float*)d_in[19];
  const float* aw0 = (const float*)d_in[20];
  const float* ab0 = (const float*)d_in[21];
  const float* aw1 = (const float*)d_in[22];
  const float* ab1 = (const float*)d_in[23];
  (void)in_sizes; (void)n_in; (void)out_size; (void)ws_size;

  const int N = GN, D = GD, N1 = GN1, N2 = GN2;
  const size_t DD = (size_t)D * D;

  char* p = (char*)d_ws;
  auto alloc = [&](size_t bytes) -> char* {
    char* r = p;
    p += (bytes + 255) & ~(size_t)255;
    return r;
  };
  float* A2F   = (float*)alloc((size_t)N * N * 4);
  float* GBIN  = (float*)alloc((size_t)N * N * 4);
  unsigned char* U8A = (unsigned char*)alloc((size_t)N * N);
  float* SUB   = (float*)alloc((size_t)N1 * N1 * 4);
  float* G1    = (float*)alloc((size_t)N1 * N1 * 4);
  float* G2    = (float*)alloc((size_t)N2 * N2 * 4);
  float* H1    = (float*)alloc((size_t)N  * D * 4);
  float* H2    = (float*)alloc((size_t)N1 * D * 4);
  float* Z     = (float*)alloc((size_t)N  * D * 4);
  float* T     = (float*)alloc((size_t)N  * D * 4);
  float* NEWH1 = (float*)alloc((size_t)N1 * D * 4);
  float* NEWH2 = (float*)alloc((size_t)N2 * D * 4);
  float* HB    = (float*)alloc((size_t)N2 * D * 4);
  float* NH    = (float*)alloc((size_t)N  * D * 4);
  float* FW    = (float*)alloc((size_t)N * 4);
  float* SW    = (float*)alloc((size_t)N * 4);
  float* SC    = (float*)alloc((size_t)N * 4);
  float* VALS0 = (float*)alloc((size_t)N1 * 4);
  int*   IDX0  = (int*)  alloc((size_t)N1 * 4);
  float* VALS1 = (float*)alloc((size_t)N2 * 4);
  int*   IDX1  = (int*)  alloc((size_t)N2 * 4);
  float* DEG   = (float*)alloc((size_t)N1 * 4);
  float* CSC   = (float*)alloc(256);

  float* out_hs0 = (float*)d_out;
  float* out_hs1 = out_hs0 + (size_t)N1 * D;
  float* out_h   = out_hs1 + (size_t)N * D;

  binarize_f32<<<ew_blocks((long)N * N), 256, 0, stream>>>(g, GBIN, (long)N * N);
  launch_gin(GBIN, h, dw1, db1, dw2, db2, nullptr, Z, T, H1, N, stream);

  rowdot<<<(N + 255) / 256, 256, 0, stream>>>(H1, pwf, pbf, FW, N, D);
  rowdot<<<(N + 255) / 256, 256, 0, stream>>>(C0, pws, pbs, SW, N, 6);
  dot_reduce<<<1, 1024, 0, stream>>>(SW, aw0, ab0, CSC, N);
  sigmoid_combine<<<(N + 255) / 256, 256, 0, stream>>>(FW, CSC, SC, N);
  topk_sort<<<1, 1024, 0, stream>>>(SC, N, N1, VALS0, IDX0);
  gather_scale<<<N1, 256, 0, stream>>>(H1, IDX0, VALS0, NEWH1, D);
  binarize_u8<<<ew_blocks((long)N * N), 256, 0, stream>>>(g, U8A, (long)N * N);
  {
    const long tiles = (long)((N + 15) >> 4) * ((N + 15) >> 4);
    adj2_iu8_wmma<<<(int)((tiles + 7) / 8), dim3(32, 8), 0, stream>>>(U8A, A2F, N);
  }
  gather_sub<<<ew_blocks((long)N1 * N1), 256, 0, stream>>>(A2F, IDX0, SUB, N, N1);
  rowsum<<<(N1 + 255) / 256, 256, 0, stream>>>(SUB, DEG, N1);
  colnorm<<<ew_blocks((long)N1 * N1), 256, 0, stream>>>(SUB, DEG, G1, N1);

  binarize_f32<<<ew_blocks((long)N1 * N1), 256, 0, stream>>>(G1, GBIN, (long)N1 * N1);
  launch_gin(GBIN, NEWH1, dw1 + DD, db1 + D, dw2 + DD, db2 + D, nullptr, Z, T, H2, N1, stream);

  rowdot<<<(N1 + 255) / 256, 256, 0, stream>>>(H2, pwf + D, pbf + 1, FW, N1, D);
  rowdot<<<(N1 + 255) / 256, 256, 0, stream>>>(C1, pws + 6, pbs + 1, SW, N1, 6);
  dot_reduce<<<1, 1024, 0, stream>>>(SW, aw1, ab1, CSC, N1);
  sigmoid_combine<<<(N1 + 255) / 256, 256, 0, stream>>>(FW, CSC, SC, N1);
  topk_sort<<<1, 1024, 0, stream>>>(SC, N1, N2, VALS1, IDX1);
  gather_scale<<<N2, 256, 0, stream>>>(H2, IDX1, VALS1, NEWH2, D);
  binarize_u8<<<ew_blocks((long)N1 * N1), 256, 0, stream>>>(G1, U8A, (long)N1 * N1);
  {
    const long tiles = (long)((N1 + 15) >> 4) * ((N1 + 15) >> 4);
    adj2_iu8_wmma<<<(int)((tiles + 7) / 8), dim3(32, 8), 0, stream>>>(U8A, A2F, N1);
  }
  gather_sub<<<ew_blocks((long)N2 * N2), 256, 0, stream>>>(A2F, IDX1, SUB, N1, N2);
  rowsum<<<(N2 + 255) / 256, 256, 0, stream>>>(SUB, DEG, N2);
  colnorm<<<ew_blocks((long)N2 * N2), 256, 0, stream>>>(SUB, DEG, G2, N2);

  binarize_f32<<<ew_blocks((long)N2 * N2), 256, 0, stream>>>(G2, GBIN, (long)N2 * N2);
  launch_gin(GBIN, NEWH2, bw1, bb1, bw2, bb2, nullptr, Z, T, HB, N2, stream);

  zero_f32<<<ew_blocks((long)N1 * D), 256, 0, stream>>>(NH, (long)N1 * D);
  scatter_rows<<<N2, 256, 0, stream>>>(HB, IDX1, NH, D);
  binarize_f32<<<ew_blocks((long)N1 * N1), 256, 0, stream>>>(G1, GBIN, (long)N1 * N1);
  launch_gin(GBIN, NH, uw1, ub1, uw2, ub2,  H2, Z, T, out_hs0, N1, stream);

  zero_f32<<<ew_blocks((long)N * D), 256, 0, stream>>>(NH, (long)N * D);
  scatter_rows<<<N1, 256, 0, stream>>>(out_hs0, IDX0, NH, D);
  binarize_f32<<<ew_blocks((long)N * N), 256, 0, stream>>>(g, GBIN, (long)N * N);
  launch_gin(GBIN, NH, uw1 + DD, ub1 + D, uw2 + DD, ub2 + D,  H1, Z, T, out_hs1, N, stream);

  add_f32<<<ew_blocks((long)N * D), 256, 0, stream>>>(out_hs1, h, out_h, (long)N * D);
}
